// FAISSAnchorKernelLayer_53171695125253
// MI455X (gfx1250) — matrix operand order repaired and checked by reading, not compiled, not run on hardware
//
#include <hip/hip_runtime.h>
#include <hip/hip_bf16.h>

typedef __attribute__((ext_vector_type(16))) _Float16 v16h;
typedef __attribute__((ext_vector_type(8)))  _Float16 v8h;
typedef __attribute__((ext_vector_type(8)))  float    v8f;

#define A_CNT   1024
#define D_OUT   256
#define H_DIM   1024
#define M_TILE  32
#define K_CHUNK 256
#define H_PAD   264   // row stride in halves: 528B, 528 % 256 == 16 -> conflict-free

// ---------- prep: w2 [1024][256] f32 -> w2t [256][1024] f16 (transposed) ----------
__global__ void prep_w2t_kernel(const float* __restrict__ w2, _Float16* __restrict__ w2t) {
    int i = blockIdx.x * 256 + threadIdx.x;      // 0 .. 262143
    int n = i >> 10;                             // output column 0..255
    int k = i & 1023;                            // hidden index 0..1023
    w2t[n * H_DIM + k] = (_Float16)w2[k * D_OUT + n];
}

// ---------- prep: Key [1024][3] -> packed (kx,ky,kz,|k|^2) float4 ----------
__global__ void prep_keys_kernel(const float* __restrict__ Key, float4* __restrict__ kp) {
    int a = blockIdx.x * 256 + threadIdx.x;      // 0 .. 1023
    float kx = Key[a * 3 + 0], ky = Key[a * 3 + 1], kz = Key[a * 3 + 2];
    float4 v; v.x = kx; v.y = ky; v.z = kz; v.w = kx * kx + ky * ky + kz * kz;
    kp[a] = v;
}

// ---------- fused: 1-NN argmin -> gelu MLP -> WMMA GEMM -> +b2 +V[idx] ----------
__global__ void __launch_bounds__(256)
faiss_anchor_fused_kernel(const float*  __restrict__ x,
                          const float4* __restrict__ keysPk,
                          const float*  __restrict__ init_mat,
                          const float*  __restrict__ Value,
                          const float*  __restrict__ w1,
                          const float*  __restrict__ b1,
                          const _Float16* __restrict__ w2t,
                          const float*  __restrict__ b2,
                          float* __restrict__ out) {
    __shared__ __align__(16) float4   sKey[A_CNT];          // 16 KB
    __shared__ __align__(16) _Float16 sH[M_TILE * H_PAD];   // 16.5 KB f16 h-chunk
    __shared__ float sX[M_TILE][3];
    __shared__ float sRL[M_TILE][3];
    __shared__ float sBest[256];
    __shared__ int   sBidx[256];
    __shared__ int   sIdx[M_TILE];

    const int tid  = threadIdx.x;
    const int lane = tid & 31;
    const int wave = tid >> 5;
    const int half = lane >> 4;       // 0: lanes 0-15, 1: lanes 16-31
    const int l15  = lane & 15;
    const int rowBase = blockIdx.x * M_TILE;

    // stage packed anchors into LDS
    #pragma unroll
    for (int i = 0; i < 4; ++i) sKey[tid + i * 256] = keysPk[tid + i * 256];
    // stage the 32 token rows (3 floats each)
    if (tid < M_TILE * 3) {
        int r = tid / 3, c = tid - r * 3;
        sX[r][c] = x[(rowBase + r) * 3 + c];
    }
    __syncthreads();

    // ---- phase 1: argmin over 1024 anchors, 8 threads per row ----
    {
        const int row  = tid & (M_TILE - 1);
        const int part = tid >> 5;                // 8 parts x 128 anchors
        const float x0 = sX[row][0], x1 = sX[row][1], x2 = sX[row][2];
        float best = 3.402823466e38f; int bidx = 0;
        const int a0 = part * 128;
        #pragma unroll 4
        for (int a = a0; a < a0 + 128; ++a) {
            float4 k = sKey[a];
            float d = k.w - 2.0f * (x0 * k.x + x1 * k.y + x2 * k.z);
            if (d < best) { best = d; bidx = a; }
        }
        sBest[tid] = best; sBidx[tid] = bidx;
    }
    __syncthreads();
    if (tid < M_TILE) {
        float best = sBest[tid]; int bidx = sBidx[tid];
        #pragma unroll
        for (int p = 1; p < 8; ++p) {
            float v = sBest[p * 32 + tid];
            if (v < best) { best = v; bidx = sBidx[p * 32 + tid]; }
        }
        sIdx[tid] = bidx;
        float4 k = sKey[bidx];
        sRL[tid][0] = sX[tid][0] - k.x;
        sRL[tid][1] = sX[tid][1] - k.y;
        sRL[tid][2] = sX[tid][2] - k.z;
    }
    __syncthreads();

    // ---- phases 2+3: loop over K in chunks of 256; accumulate WMMA tiles ----
    const int rt = wave >> 2;                 // fixed 16-row tile per wave
    const int ctBase = (wave & 3) * 4;        // 4 consecutive 16-col tiles per wave
    v8f acc0 = {}, acc1 = {}, acc2 = {}, acc3 = {};

    const _Float16* hrow = &sH[(rt * 16 + l15) * H_PAD];

    for (int kc = 0; kc < H_DIM / K_CHUNK; ++kc) {
        // phase 2: one gelu column of h per thread, all 32 rows
        const int kg = kc * K_CHUNK + tid;
        const float wa = w1[kg], wb = w1[H_DIM + kg], wc = w1[2 * H_DIM + kg];
        const float bb = b1[kg];
        #pragma unroll 4
        for (int m = 0; m < M_TILE; ++m) {
            float pre = sRL[m][0] * wa + sRL[m][1] * wb + sRL[m][2] * wc + bb;
            float g = 0.5f * pre * (1.0f + erff(pre * 0.70710678118654752f));
            sH[m * H_PAD + tid] = (_Float16)g;
        }
        __syncthreads();

        // phase 3: 8 x (k=32) WMMA steps over this chunk
        #pragma unroll
        for (int kk = 0; kk < 8; ++kk) {
            const int q = kk * 32;
            // A fragment: contiguous K layout (elems 0-15: K=half*16+e), matching B
            v8h lo = *(const v8h*)&hrow[q + half * 16];
            v8h hi = *(const v8h*)&hrow[q + half * 16 + 8];
            v16h afrag;
            #pragma unroll
            for (int i = 0; i < 8; ++i) { afrag[i] = lo[i]; afrag[i + 8] = hi[i]; }

            // B fragments: lane holds col N=l15, K = half*16 + e  -> contiguous in w2t[N][K]
            const int kB = kc * K_CHUNK + q + half * 16;
            v16h b0 = *(const v16h*)&w2t[(ctBase * 16 + 0 * 16 + l15) * H_DIM + kB];
            v16h b1f = *(const v16h*)&w2t[(ctBase * 16 + 1 * 16 + l15) * H_DIM + kB];
            v16h b2f = *(const v16h*)&w2t[(ctBase * 16 + 2 * 16 + l15) * H_DIM + kB];
            v16h b3f = *(const v16h*)&w2t[(ctBase * 16 + 3 * 16 + l15) * H_DIM + kB];

            acc0 = __builtin_amdgcn_wmma_f32_16x16x32_f16(false, afrag, false, b0,  (short)0, acc0, false, false);
            acc1 = __builtin_amdgcn_wmma_f32_16x16x32_f16(false, afrag, false, b1f, (short)0, acc1, false, false);
            acc2 = __builtin_amdgcn_wmma_f32_16x16x32_f16(false, afrag, false, b2f, (short)0, acc2, false, false);
            acc3 = __builtin_amdgcn_wmma_f32_16x16x32_f16(false, afrag, false, b3f, (short)0, acc3, false, false);
        }
        __syncthreads();
    }

    // ---- phase 4: epilogue, fused +b2 + (init_mat+Value)[idx], C/D layout ----
    v8f accs[4] = { acc0, acc1, acc2, acc3 };
    #pragma unroll
    for (int s = 0; s < 4; ++s) {
        const int n = (ctBase + s) * 16 + l15;
        const float bias = b2[n];
        #pragma unroll
        for (int r = 0; r < 8; ++r) {
            const int mloc = rt * 16 + r + 8 * half;     // D layout: M = r + 8*(lane>=16)
            const int row  = rowBase + mloc;
            const int a    = sIdx[mloc];
            float v = accs[s][r] + bias + init_mat[a * D_OUT + n] + Value[a * D_OUT + n];
            out[row * D_OUT + n] = v;
        }
    }
}

extern "C" void kernel_launch(void* const* d_in, const int* in_sizes, int n_in,
                              void* d_out, int out_size, void* d_ws, size_t ws_size,
                              hipStream_t stream) {
    const float* x        = (const float*)d_in[0];
    const float* Key      = (const float*)d_in[1];
    const float* init_mat = (const float*)d_in[2];
    const float* Value    = (const float*)d_in[3];
    const float* w1       = (const float*)d_in[4];
    const float* b1       = (const float*)d_in[5];
    const float* w2       = (const float*)d_in[6];
    const float* b2       = (const float*)d_in[7];
    float* out = (float*)d_out;

    _Float16* w2t  = (_Float16*)d_ws;                          // 512 KB
    float4*  keysP = (float4*)((char*)d_ws + 256 * 1024 * 2);  // 16 KB

    prep_w2t_kernel <<<1024, 256, 0, stream>>>(w2, w2t);
    prep_keys_kernel<<<4,    256, 0, stream>>>(Key, keysP);

    const int nRows = 8 * 8192;                                // B*N = 65536
    faiss_anchor_fused_kernel<<<nRows / M_TILE, 256, 0, stream>>>(
        x, keysP, init_mat, Value, w1, b1, w2t, b2, out);
}
